// NeRF_13838384628262
// MI455X (gfx1250) — compile-verified, matrix operand order checked by reading, not run on hardware
//
#include <hip/hip_runtime.h>

#define N_RAYS   2048
#define N_COARSE 64
#define N_FINE   128
#define N_TOTAL  (N_COARSE + N_FINE)

// 64 points per block, 8 wave32s; each wave owns N-tiles {wave, wave+8} and
// all 4 M-tiles. A-fragments loaded once per K-chunk and shared across both
// N-tiles; each B-fragment feeds 4 WMMAs (one per M-tile).
#define MPTS 64

typedef __attribute__((ext_vector_type(16))) _Float16 v16h;
typedef __attribute__((ext_vector_type(8)))  float    v8f;

struct MlpW {
  const _Float16* Wp[10];   // packed f16 fragment-order weights W0..W9
  const float*    b[10];    // biases b0..b9 (f32, original)
  const float* Wsig; const float* bsig;
  const float* Wrgb; const float* brgb;
};

// ---------------------------------------------------------------------------
// Weight repack: W (f32, din x dout, row-major) -> f16 WMMA-B fragments.
// Packed index = (((tile*nkc + kc)*32 + lane)*16 + h).
// B fragment layout (mirror of 16-bit A layout, N in the role of M):
//   lane<16 : n = lane,     koff = (h<8 ? h : h+8)
//   lane>=16: n = lane-16,  koff = (h<8 ? h : h+8) + 8
//   k = kc*32 + koff ; zero-fill k >= din.
// ---------------------------------------------------------------------------
__global__ void pack_weights(const float* __restrict__ W, _Float16* __restrict__ out,
                             int din, int dout, int nkc) {
  size_t total = (size_t)(dout / 16) * nkc * 32 * 16;
  for (size_t idx = (size_t)blockIdx.x * blockDim.x + threadIdx.x; idx < total;
       idx += (size_t)gridDim.x * blockDim.x) {
    int h    = (int)(idx & 15);
    int lane = (int)((idx >> 4) & 31);
    size_t rest = idx >> 9;
    int kc   = (int)(rest % (size_t)nkc);
    int tile = (int)(rest / (size_t)nkc);
    int n = tile * 16 + (lane & 15);
    int k = kc * 32 + ((lane & 16) ? 8 : 0) + (h < 8 ? h : h + 8);
    float v = (k < din) ? W[(size_t)k * dout + n] : 0.0f;
    out[idx] = (_Float16)v;
  }
}

// t_c = near + (far - near) * s/(N_COARSE-1)
__global__ void gen_tc(const float* __restrict__ near_, const float* __restrict__ far_,
                       float* __restrict__ t_c) {
  int i = blockIdx.x * blockDim.x + threadIdx.x;
  if (i >= N_RAYS * N_COARSE) return;
  int ray = i / N_COARSE, s = i - ray * N_COARSE;
  float f = (float)s / (float)(N_COARSE - 1);
  t_c[i] = near_[ray] + (far_[ray] - near_[ray]) * f;
}

// ---------------------------------------------------------------------------
// A-fragment (16-bit A matrix, 16x32) from LDS activation buffer (ld = 320).
// lane<16 : M = lane,    halves 0..7 -> K = kb+0..7,  halves 8..15 -> K = kb+16..23
// lane>=16: M = lane-16, same pattern with kb += 8.
// ---------------------------------------------------------------------------
__device__ __forceinline__ v16h load_A_frag(const _Float16* in, int lane, int kc) {
  int m  = lane & 15;
  int kb = kc * 32 + ((lane & 16) ? 8 : 0);
  const _Float16* r = in + m * 320 + kb;
  v16h a;
#pragma unroll
  for (int j = 0; j < 8; ++j) { a[j] = r[j]; a[j + 8] = r[16 + j]; }
  return a;
}

// One dense layer: D(64 x 16*NTILES) = A(64xK) x B(K x 16*NTILES) + bias.
// kc outer: 4 A-fragments loaded once, shared by both N-tiles (8 WMMAs/kc).
template<int NKC, int NTILES, bool RELU>
__device__ __forceinline__ void wmma_layer(const _Float16* __restrict__ in,
                                           const _Float16* __restrict__ Wp,
                                           const float* __restrict__ bias,
                                           _Float16* __restrict__ out,
                                           int wave, int lane) {
  v8f acc[NTILES][4];
  v8f zero = {};
#pragma unroll
  for (int tt = 0; tt < NTILES; ++tt)
#pragma unroll
    for (int mt = 0; mt < 4; ++mt) acc[tt][mt] = zero;

#pragma unroll
  for (int kc = 0; kc < NKC; ++kc) {
    v16h a0 = load_A_frag(in,            lane, kc);
    v16h a1 = load_A_frag(in + 16 * 320, lane, kc);
    v16h a2 = load_A_frag(in + 32 * 320, lane, kc);
    v16h a3 = load_A_frag(in + 48 * 320, lane, kc);
#pragma unroll
    for (int tt = 0; tt < NTILES; ++tt) {
      const _Float16* wp = Wp + ((size_t)(wave + 8 * tt) * NKC + kc) * 512;
      v16h b = *(const v16h*)(wp + (size_t)lane * 16);
      acc[tt][0] = __builtin_amdgcn_wmma_f32_16x16x32_f16(false, a0, false, b, (short)0, acc[tt][0], false, false);
      acc[tt][1] = __builtin_amdgcn_wmma_f32_16x16x32_f16(false, a1, false, b, (short)0, acc[tt][1], false, false);
      acc[tt][2] = __builtin_amdgcn_wmma_f32_16x16x32_f16(false, a2, false, b, (short)0, acc[tt][2], false, false);
      acc[tt][3] = __builtin_amdgcn_wmma_f32_16x16x32_f16(false, a3, false, b, (short)0, acc[tt][3], false, false);
    }
  }
  // C layout: lane<16 -> N=lane, reg r -> M=r ; lane>=16 -> N=lane-16, M=r+8
#pragma unroll
  for (int tt = 0; tt < NTILES; ++tt) {
    const int tile = wave + 8 * tt;
    const int n  = tile * 16 + (lane & 15);
    const int mb = (lane & 16) ? 8 : 0;
    const float bi = bias[n];
#pragma unroll
    for (int mt = 0; mt < 4; ++mt) {
#pragma unroll
      for (int r = 0; r < 8; ++r) {
        float v = acc[tt][mt][r] + bi;
        if (RELU) v = fmaxf(v, 0.0f);
        out[(mt * 16 + mb + r) * 320 + n] = (_Float16)v;
      }
    }
  }
}

// ---------------------------------------------------------------------------
// Fused NeRF MLP: 64 points per block. enc_x parked in buf1[:,256:320]
// (layer writes touch cols 0..255 only, so it survives until skip layer L5);
// enc_d parked in buf0[:,288:320] and copied to buf1[:,256:288] before L9.
// ---------------------------------------------------------------------------
__global__ __launch_bounds__(256)
void nerf_mlp(const float* __restrict__ rays_o, const float* __restrict__ rays_d,
              const float* __restrict__ tvals, int S, MlpW w,
              float* __restrict__ sigma_out, float* __restrict__ rgb_out) {
  extern __shared__ __align__(16) char smem_raw[];
  _Float16* buf0 = (_Float16*)smem_raw;            // MPTS*320
  _Float16* buf1 = buf0 + MPTS * 320;              // MPTS*320
  float* xyz  = (float*)(buf1 + MPTS * 320);       // MPTS*3
  float* dirv = xyz + MPTS * 3;                    // MPTS*3
  float* red  = dirv + MPTS * 3;                   // 256 (sigma reduction)

  const int tid  = threadIdx.x;
  const int wave = tid >> 5, lane = tid & 31;
  const int p0   = blockIdx.x * MPTS;
  const float PI = 3.14159265358979323846f;

  if (tid < MPTS * 3) {
    int m = tid / 3, c = tid - m * 3;
    int p = p0 + m;
    int ray = p / S, s = p - ray * S;
    float t = tvals[(size_t)ray * S + s];
    float d = rays_d[ray * 3 + c];
    xyz[m * 3 + c]  = rays_o[ray * 3 + c] + d * t;
    dirv[m * 3 + c] = d;
  }
  __syncthreads();

  // enc_x: [x(3) | c-major: 10 freqs x (sin,cos)] = 63 dims, pad to 64 (zero)
  for (int idx = tid; idx < MPTS * 64; idx += 256) {
    int m = idx >> 6, j = idx & 63;
    float v = 0.0f;
    if (j < 3) v = xyz[m * 3 + j];
    else if (j < 63) {
      int q = j - 3; int c = q / 20; int r = q - c * 20;
      float ang = xyz[m * 3 + c] * (PI * (float)(1 << (r >> 1)));
      v = (r & 1) ? __cosf(ang) : __sinf(ang);
    }
    _Float16 h = (_Float16)v;
    buf0[m * 320 + j] = h;          // layer-0 input (K padded to 64)
    buf1[m * 320 + 256 + j] = h;    // survives until skip layer L5
  }
  // enc_d: 27 dims, pad to 32 (zero), parked in buf0 cols 288..319
  for (int idx = tid; idx < MPTS * 32; idx += 256) {
    int m = idx >> 5, j = idx & 31;
    float v = 0.0f;
    if (j < 3) v = dirv[m * 3 + j];
    else if (j < 27) {
      int q = j - 3; int c = q / 8; int r = q - c * 8;
      float ang = dirv[m * 3 + c] * (PI * (float)(1 << (r >> 1)));
      v = (r & 1) ? __cosf(ang) : __sinf(ang);
    }
    buf0[m * 320 + 288 + j] = (_Float16)v;
  }
  __syncthreads();

  wmma_layer<2,  2, true>(buf0, w.Wp[0], w.b[0], buf1, wave, lane); __syncthreads();
  wmma_layer<8,  2, true>(buf1, w.Wp[1], w.b[1], buf0, wave, lane); __syncthreads();
  wmma_layer<8,  2, true>(buf0, w.Wp[2], w.b[2], buf1, wave, lane); __syncthreads();
  wmma_layer<8,  2, true>(buf1, w.Wp[3], w.b[3], buf0, wave, lane); __syncthreads();
  wmma_layer<8,  2, true>(buf0, w.Wp[4], w.b[4], buf1, wave, lane); __syncthreads();
  // buf1 now = [h(256) | enc_x(63) | 0] -> 320-wide skip input
  wmma_layer<10, 2, true>(buf1, w.Wp[5], w.b[5], buf0, wave, lane); __syncthreads();
  wmma_layer<8,  2, true>(buf0, w.Wp[6], w.b[6], buf1, wave, lane); __syncthreads();
  wmma_layer<8,  2, true>(buf1, w.Wp[7], w.b[7], buf0, wave, lane); __syncthreads();

  // sigma head: relu(h . Wsig + bsig); 4 partial sums per point + reduction
  {
    int m = tid >> 2, q = tid & 3;
    const _Float16* hrow = buf0 + m * 320 + q * 64;
    const float* ws = w.Wsig + q * 64;
    float s = 0.0f;
#pragma unroll 1
    for (int k = 0; k < 64; ++k) s += (float)hrow[k] * ws[k];
    red[tid] = s;
  }
  __syncthreads();
  if (tid < MPTS) {
    float s = red[tid * 4] + red[tid * 4 + 1] + red[tid * 4 + 2] + red[tid * 4 + 3]
            + w.bsig[0];
    sigma_out[p0 + tid] = fmaxf(s, 0.0f);
  }
  // bottleneck (no relu): buf0 -> buf1
  wmma_layer<8, 2, false>(buf0, w.Wp[8], w.b[8], buf1, wave, lane);
  __syncthreads();
  // concat enc_d: buf0[:,288:320] -> buf1[:,256:288]   (288-wide, nkc = 9)
  for (int idx = tid; idx < MPTS * 32; idx += 256) {
    int m = idx >> 5, j = idx & 31;
    buf1[m * 320 + 256 + j] = buf0[m * 320 + 288 + j];
  }
  __syncthreads();
  wmma_layer<9, 1, true>(buf1, w.Wp[9], w.b[9], buf0, wave, lane);  // 128-wide
  __syncthreads();

  // rgb head: sigmoid(h2 . Wrgb + brgb)
  if (tid < MPTS * 3) {
    int m = tid / 3, c = tid - m * 3;
    float s = 0.0f;
#pragma unroll 1
    for (int k = 0; k < 128; ++k) s += (float)buf0[m * 320 + k] * w.Wrgb[k * 3 + c];
    s += w.brgb[c];
    rgb_out[(size_t)(p0 + m) * 3 + c] = 1.0f / (1.0f + __expf(-s));
  }
}

// ---------------------------------------------------------------------------
// Per-ray coarse compositing + inverse-CDF resampling + sorted merge -> t_f.
// ---------------------------------------------------------------------------
__global__ void composite_coarse(const float* __restrict__ sigma,
                                 const float* __restrict__ rgb,
                                 const float* __restrict__ t_c,
                                 float* __restrict__ out,      // d_out, writes (0, ray, :)
                                 float* __restrict__ t_f) {
  int ray = blockIdx.x * blockDim.x + threadIdx.x;
  if (ray >= N_RAYS) return;
  float t[N_COARSE], wgt[N_COARSE];
  for (int i = 0; i < N_COARSE; ++i) t[i] = t_c[(size_t)ray * N_COARSE + i];
  float T = 1.0f, acc = 0.0f, r0 = 0.0f, r1 = 0.0f, r2 = 0.0f;
  for (int i = 0; i < N_COARSE; ++i) {
    float dist = (i < N_COARSE - 1) ? (t[i + 1] - t[i]) : 1e10f;
    float a = 1.0f - __expf(-sigma[(size_t)ray * N_COARSE + i] * dist);
    float wi = a * T;
    T *= (1.0f - a + 1e-10f);
    wgt[i] = wi; acc += wi;
    const float* c = rgb + (size_t)(ray * N_COARSE + i) * 3;
    r0 += wi * c[0]; r1 += wi * c[1]; r2 += wi * c[2];
  }
  float bg = 1.0f - acc;
  out[ray * 3 + 0] = r0 + bg; out[ray * 3 + 1] = r1 + bg; out[ray * 3 + 2] = r2 + bg;

  // sample_pdf (deterministic u = linspace(0,1,128))
  float bins[N_COARSE - 1], cdf[N_COARSE - 1], pdf[N_COARSE - 2];
  for (int i = 0; i < N_COARSE - 1; ++i) bins[i] = 0.5f * (t[i + 1] + t[i]);
  float wsum = 0.0f;
  for (int j = 0; j < N_COARSE - 2; ++j) { pdf[j] = wgt[j + 1] + 1e-5f; wsum += pdf[j]; }
  float inv = 1.0f / wsum;
  cdf[0] = 0.0f;
  for (int j = 0; j < N_COARSE - 2; ++j) cdf[j + 1] = cdf[j] + pdf[j] * inv;
  float smp[N_FINE];
  for (int f = 0; f < N_FINE; ++f) {
    float u = (float)f / (float)(N_FINE - 1);
    int idx = 0;
    while (idx < N_COARSE - 1 && cdf[idx] <= u) idx++;   // searchsorted right
    int below = (idx - 1 < 0) ? 0 : idx - 1;
    int above = (idx > N_COARSE - 2) ? N_COARSE - 2 : idx;
    float cb = cdf[below], ca = cdf[above];
    float den = (ca - cb < 1e-5f) ? 1.0f : (ca - cb);
    smp[f] = bins[below] + (u - cb) / den * (bins[above] - bins[below]);
  }
  // merge two sorted lists (t_c asc, smp asc) -> t_f (192)
  int i = 0, j = 0;
  float* dst = t_f + (size_t)ray * N_TOTAL;
  for (int k = 0; k < N_TOTAL; ++k) {
    bool takeT = (j >= N_FINE) || (i < N_COARSE && t[i] <= smp[j]);
    dst[k] = takeT ? t[i++] : smp[j++];
  }
}

__global__ void composite_fine(const float* __restrict__ sigma,
                               const float* __restrict__ rgb,
                               const float* __restrict__ t_f,
                               float* __restrict__ out) {  // writes (1, ray, :)
  int ray = blockIdx.x * blockDim.x + threadIdx.x;
  if (ray >= N_RAYS) return;
  const float* tp = t_f + (size_t)ray * N_TOTAL;
  float T = 1.0f, acc = 0.0f, r0 = 0.0f, r1 = 0.0f, r2 = 0.0f;
  float tc = tp[0];
  for (int i = 0; i < N_TOTAL; ++i) {
    float tn = (i < N_TOTAL - 1) ? tp[i + 1] : 0.0f;
    float dist = (i < N_TOTAL - 1) ? (tn - tc) : 1e10f;
    float a = 1.0f - __expf(-sigma[(size_t)ray * N_TOTAL + i] * dist);
    float wi = a * T;
    T *= (1.0f - a + 1e-10f);
    acc += wi;
    const float* c = rgb + (size_t)(ray * N_TOTAL + i) * 3;
    r0 += wi * c[0]; r1 += wi * c[1]; r2 += wi * c[2];
    tc = tn;
  }
  float bg = 1.0f - acc;
  out[(N_RAYS + ray) * 3 + 0] = r0 + bg;
  out[(N_RAYS + ray) * 3 + 1] = r1 + bg;
  out[(N_RAYS + ray) * 3 + 2] = r2 + bg;
}

// ---------------------------------------------------------------------------
// Input layout (JAX pytree: dicts flatten in sorted-key order):
//   d_in[0..3]  : rays_o, rays_d, near, far
//   d_in[4..27] : coarse  W0..W9, Wrgb, Wsig, b0..b9, brgb, bsig
//   d_in[28..51]: fine    (same order)
// ---------------------------------------------------------------------------
extern "C" void kernel_launch(void* const* d_in, const int* in_sizes, int n_in,
                              void* d_out, int out_size, void* d_ws, size_t ws_size,
                              hipStream_t stream) {
  (void)in_sizes; (void)n_in; (void)out_size; (void)ws_size;
  const float* rays_o = (const float*)d_in[0];
  const float* rays_d = (const float*)d_in[1];
  const float* near_  = (const float*)d_in[2];
  const float* far_   = (const float*)d_in[3];

  static const int din_[10]  = {63, 256, 256, 256, 256, 319, 256, 256, 256, 283};
  static const int dout_[10] = {256, 256, 256, 256, 256, 256, 256, 256, 256, 128};
  size_t off[11]; off[0] = 0;
  for (int i = 0; i < 10; ++i) {
    int nkc = (din_[i] + 31) / 32;
    off[i + 1] = off[i] + (size_t)(dout_[i] / 16) * nkc * 512;
  }
  const size_t packHalves = off[10];   // 593,920 halves per MLP

  _Float16* packC = (_Float16*)d_ws;
  _Float16* packF = packC + packHalves;
  float* fb    = (float*)(packF + packHalves);
  float* t_c   = fb; fb += (size_t)N_RAYS * N_COARSE;
  float* sig_c = fb; fb += (size_t)N_RAYS * N_COARSE;
  float* rgb_c = fb; fb += (size_t)N_RAYS * N_COARSE * 3;
  float* t_f   = fb; fb += (size_t)N_RAYS * N_TOTAL;
  float* sig_f = fb; fb += (size_t)N_RAYS * N_TOTAL;
  float* rgb_f = fb; fb += (size_t)N_RAYS * N_TOTAL * 3;

  MlpW wmlp[2];
  for (int mlp = 0; mlp < 2; ++mlp) {
    int base = 4 + mlp * 24;
    _Float16* pack = mlp ? packF : packC;
    for (int i = 0; i < 10; ++i) {
      int nkc = (din_[i] + 31) / 32;
      size_t total = (size_t)(dout_[i] / 16) * nkc * 512;
      pack_weights<<<(int)((total + 255) / 256), 256, 0, stream>>>(
          (const float*)d_in[base + i], pack + off[i], din_[i], dout_[i], nkc);
      wmlp[mlp].Wp[i] = pack + off[i];
      wmlp[mlp].b[i]  = (const float*)d_in[base + 12 + i];  // b0..b7,b8,b9 contiguous
    }
    wmlp[mlp].Wrgb = (const float*)d_in[base + 10];
    wmlp[mlp].Wsig = (const float*)d_in[base + 11];
    wmlp[mlp].brgb = (const float*)d_in[base + 22];
    wmlp[mlp].bsig = (const float*)d_in[base + 23];
  }

  gen_tc<<<(N_RAYS * N_COARSE + 255) / 256, 256, 0, stream>>>(near_, far_, t_c);

  const size_t smemBytes = (size_t)2 * MPTS * 320 * sizeof(_Float16)
                         + (size_t)(MPTS * 3 * 2 + 256) * sizeof(float);

  nerf_mlp<<<N_RAYS * N_COARSE / MPTS, 256, smemBytes, stream>>>(
      rays_o, rays_d, t_c, N_COARSE, wmlp[0], sig_c, rgb_c);

  composite_coarse<<<(N_RAYS + 255) / 256, 256, 0, stream>>>(
      sig_c, rgb_c, t_c, (float*)d_out, t_f);

  nerf_mlp<<<N_RAYS * N_TOTAL / MPTS, 256, smemBytes, stream>>>(
      rays_o, rays_d, t_f, N_TOTAL, wmlp[1], sig_f, rgb_f);

  composite_fine<<<(N_RAYS + 255) / 256, 256, 0, stream>>>(
      sig_f, rgb_f, t_f, (float*)d_out);
}
